// GRU_75935021794059
// MI455X (gfx1250) — matrix operand order repaired and checked by reading, not compiled, not run on hardware
//
#include <hip/hip_runtime.h>
#include <math.h>

typedef _Float16 f16;
typedef __attribute__((ext_vector_type(8)))  _Float16 v8h;
typedef __attribute__((ext_vector_type(16))) _Float16 v16h;
typedef __attribute__((ext_vector_type(8)))  float    v8f;
typedef __attribute__((ext_vector_type(4)))  unsigned int v4u;
typedef __attribute__((ext_vector_type(8)))  int v8i;
typedef __attribute__((ext_vector_type(4)))  int v4i;

namespace {
constexpr int kB = 256;   // batch
constexpr int kT = 512;   // timesteps
constexpr int kD = 512;   // input dim
constexpr int kH = 1024;  // hidden dim
constexpr int kKZR = kD + kH;      // 1536: K of fused [x_t | h] GEMM
constexpr int BM = 64, BN = 64, BK = 32;
constexpr int NCHUNK = kKZR / BK;  // 48 K-chunks
constexpr int NTHREADS = 128;      // 4 wave32 waves
}

__device__ __forceinline__ float sigmoidf_(float v) {
  return 1.0f / (1.0f + __expf(-v));
}

// Issue a TDM 2D tile load: BN x BK f16 tile of WpT (row-major [NT, kKZR]) into LDS
// at lds_off, with 16B padding after each 64B row (-> LDS row stride BK+8 f16 = 80B).
__device__ __forceinline__ void tdm_load_b(const f16* gsrc, unsigned int lds_off, int NT) {
  const unsigned long long ga = (unsigned long long)(uintptr_t)gsrc;
  v4u g0;
  g0[0] = 1u;                                        // count=1 (valid user descriptor)
  g0[1] = lds_off;                                   // LDS byte address
  g0[2] = (unsigned int)(ga & 0xFFFFFFFFu);          // global_addr[31:0]
  g0[3] = (unsigned int)((ga >> 32) & 0x01FFFFFFu)   // global_addr[56:32]
        | 0x80000000u;                               // type=2 ("image")
  v8i g1;
  g1[0] = (1 << 16)      // data_size = 2 bytes
        | (1 << 20)      // pad_enable
        | (3 << 22)      // pad_interval: 16 DWORDs (=64B, one tile row)
        | (3 << 25);     // pad_amount: 4 DWORDs (=16B)
  g1[1] = (int)((unsigned)(kKZR & 0xFFFF) << 16);    // tensor_dim0[15:0] @ bit 48
  g1[2] = (int)(((unsigned)kKZR >> 16) & 0xFFFFu)    // tensor_dim0[31:16]
        | (int)((unsigned)(NT & 0xFFFF) << 16);      // tensor_dim1[15:0]
  g1[3] = (int)(((unsigned)NT >> 16) & 0xFFFFu)      // tensor_dim1[31:16]
        | (int)(32u << 16);                          // tile_dim0 = 32 (k extent)
  g1[4] = 64;                                        // tile_dim1 = 64 (n rows), tile_dim2=0
  g1[5] = kKZR;                                      // tensor_dim0_stride[31:0]
  g1[6] = 0;                                         // stride hi / dim1_stride lo
  g1[7] = 0;
  const v4i gz4 = {0, 0, 0, 0};                      // 2D tensor: groups 2/3 unused
  const v8i gz8 = {0, 0, 0, 0, 0, 0, 0, 0};         // extra group (6-arg toolchain), unused
  __builtin_amdgcn_tensor_load_to_lds(g0, g1, gz4, gz4, gz8, 0);
}

// ---------------- weight packing (f32 -> f16, TRANSPOSED: [N][K], k-contiguous) ----
__global__ void pack_zr_kernel(const float* __restrict__ Wz, const float* __restrict__ Uz,
                               const float* __restrict__ Wr, const float* __restrict__ Ur,
                               f16* __restrict__ out) {
  int idx = blockIdx.x * blockDim.x + threadIdx.x;
  if (idx >= 2 * kH * kKZR) return;
  int n = idx / kKZR;
  int k = idx - n * kKZR;
  int gate = n >> 10;          // 0: z, 1: r
  int j = n & (kH - 1);
  float v;
  if (k < kD) v = gate ? Wr[(size_t)k * kH + j] : Wz[(size_t)k * kH + j];
  else        v = gate ? Ur[(size_t)(k - kD) * kH + j] : Uz[(size_t)(k - kD) * kH + j];
  out[idx] = (f16)v;
}

__global__ void pack_c_kernel(const float* __restrict__ W, const float* __restrict__ U,
                              f16* __restrict__ out) {
  int idx = blockIdx.x * blockDim.x + threadIdx.x;
  if (idx >= kH * kKZR) return;
  int n = idx / kKZR;
  int k = idx - n * kKZR;
  float v = (k < kD) ? W[(size_t)k * kH + n] : U[(size_t)(k - kD) * kH + n];
  out[idx] = (f16)v;
}

__global__ void init_h_kernel(float* __restrict__ hf, f16* __restrict__ h16) {
  int idx = blockIdx.x * blockDim.x + threadIdx.x;
  if (idx < kB * kH) { hf[idx] = 0.0f; h16[idx] = (f16)0.0f; }
}

__global__ void copy_out_kernel(const float* __restrict__ hf, float* __restrict__ out) {
  int idx = blockIdx.x * blockDim.x + threadIdx.x;
  if (idx < kB * kH) out[idx] = hf[idx];
}

// ---------------- fused GRU step GEMM (double-buffered; TDM feeds B tiles) -------
// PHASE 0: [x_t | h]   (256 x 1536) @ Wzr (1536 x 2048) -> z (sigmoid), rh16 = sigmoid(r)*h
// PHASE 1: [x_t | r*h] (256 x 1536) @ Wc  (1536 x 1024) -> h = (1-z)h + z*tanh(.)
template <int PHASE>
__global__ __launch_bounds__(NTHREADS)
void gru_step_kernel(const float* __restrict__ x, int t,
                     const f16* __restrict__ WpT,   // packed f16 weights, TRANSPOSED [NT, kKZR]
                     const f16* __restrict__ a16,   // A rows for k>=kD (h16 / rh16)
                     float* __restrict__ hf,        // f32 hidden state [kB, kH]
                     float* __restrict__ zb,        // z gate [kB, kH]
                     f16*   __restrict__ o16,       // rh16 (PHASE 0) / h16 (PHASE 1)
                     const float* __restrict__ bias0,
                     const float* __restrict__ bias1) {
  constexpr int NT = (PHASE == 0) ? 2 * kH : kH;
  __shared__ f16 sA [2][BM][BK];       // [buf][m][k], k contiguous
  __shared__ f16 sBt[2][BN][BK + 8];   // [buf][n][k], k contiguous; stride 80B = TDM pad

  const int tid  = threadIdx.x;
  const int lane = tid & 31;
  const int wv   = tid >> 5;       // wave 0..3, each owns a 16-row strip
  const int bm0  = blockIdx.y * BM;
  const int bn0  = blockIdx.x * BN;

  v8f acc[4] = {};

  // A-tile cooperative load: 64x32 f16, 16 elems/thread (all vectorized)
  const int ar   = tid >> 1;            // row 0..63
  const int ak0  = (tid & 1) * 16;      // k half: 0 or 16
  const int am   = bm0 + ar;            // batch row
  const float* xrow = x + ((size_t)am * kT + t) * kD;  // x[am, t, :]

  const f16* wtile = WpT + (size_t)bn0 * kKZR;   // B-tile column block base
  const unsigned int ldsB[2] = {
      (unsigned int)(uintptr_t)&sBt[0][0][0],
      (unsigned int)(uintptr_t)&sBt[1][0][0]};

  v8h a0r, a1r;  // register staging for next A chunk
  auto load_a = [&](int kk) {
    if (kk < kD) {                       // uniform branch per chunk
      const float* src = xrow + kk + ak0;
      float4 f0 = ((const float4*)src)[0];
      float4 f1 = ((const float4*)src)[1];
      float4 f2 = ((const float4*)src)[2];
      float4 f3 = ((const float4*)src)[3];
      if (kk + BK < kD) __builtin_prefetch(src + BK, 0, 1);   // global_prefetch_b8
      a0r = (v8h){(f16)f0.x, (f16)f0.y, (f16)f0.z, (f16)f0.w,
                  (f16)f1.x, (f16)f1.y, (f16)f1.z, (f16)f1.w};
      a1r = (v8h){(f16)f2.x, (f16)f2.y, (f16)f2.z, (f16)f2.w,
                  (f16)f3.x, (f16)f3.y, (f16)f3.z, (f16)f3.w};
    } else {
      const v8h* src = (const v8h*)&a16[(size_t)am * kH + (kk - kD) + ak0];
      a0r = src[0];
      a1r = src[1];
    }
  };
  auto store_a = [&](int buf) {
    *(v8h*)&sA[buf][ar][ak0]     = a0r;
    *(v8h*)&sA[buf][ar][ak0 + 8] = a1r;
  };

  // Prologue: fill buffer 0 (TDM for B, VALU path for A)
  if (wv == 0) tdm_load_b(wtile, ldsB[0], NT);
  load_a(0);
  store_a(0);
  if (wv == 0) __builtin_amdgcn_s_wait_tensorcnt(0);
  __syncthreads();

  const int arow = wv * 16 + (lane & 15);
  const int koff = (lane < 16) ? 0 : 16;  // A frag: K 0-15 vs 16-31
  const int kb   = (lane < 16) ? 0 : 16;  // B frag: K 0-15 vs 16-31

  #pragma unroll 2
  for (int i = 0; i < NCHUNK; ++i) {
    const int cur = i & 1;
    const bool more = (i + 1 < NCHUNK);
    if (more) {
      if (wv == 0) tdm_load_b(wtile + (i + 1) * BK, ldsB[cur ^ 1], NT);  // DMA overlaps WMMAs
      load_a((i + 1) * BK);                                              // global loads overlap
    }

    // A fragment (contiguous k = 16h + s, same order as B), 2x ds_load_b128
    v8h a0 = *(const v8h*)&sA[cur][arow][koff];
    v8h a1 = *(const v8h*)&sA[cur][arow][koff + 8];
    v16h afrag = __builtin_shufflevector(a0, a1,
        0, 1, 2, 3, 4, 5, 6, 7, 8, 9, 10, 11, 12, 13, 14, 15);

    // Prefetch all 4 B fragments (8x ds_load_b128), then run WMMAs back-to-back.
    v16h bfrag[4];
    #pragma unroll
    for (int nt = 0; nt < 4; ++nt) {
      const int ncol = nt * 16 + (lane & 15);
      v8h b0 = *(const v8h*)&sBt[cur][ncol][kb];
      v8h b1 = *(const v8h*)&sBt[cur][ncol][kb + 8];
      bfrag[nt] = __builtin_shufflevector(b0, b1,
          0, 1, 2, 3, 4, 5, 6, 7, 8, 9, 10, 11, 12, 13, 14, 15);
    }
    #pragma unroll
    for (int nt = 0; nt < 4; ++nt) {
      acc[nt] = __builtin_amdgcn_wmma_f32_16x16x32_f16(
          false, afrag, false, bfrag[nt], (short)0, acc[nt], false, false);
    }

    if (more) {
      store_a(cur ^ 1);                            // buf cur^1 last read in iter i-1 (fenced)
      if (wv == 0) __builtin_amdgcn_s_wait_tensorcnt(0);
    }
    __syncthreads();
  }

  // Epilogue. C/D layout: lanes 0-15 hold M 0..7 (VGPR r8), N=lane;
  //                       lanes 16-31 hold M 8..15, N=lane-16.
  #pragma unroll
  for (int nt = 0; nt < 4; ++nt) {
    #pragma unroll
    for (int r8 = 0; r8 < 8; ++r8) {
      const int m = bm0 + wv * 16 + ((lane < 16) ? r8 : r8 + 8);
      const int n = bn0 + nt * 16 + (lane & 15);
      const float c = acc[nt][r8];
      if constexpr (PHASE == 0) {
        const int gate = n >> 10;
        const int j = n & (kH - 1);
        const size_t o = (size_t)m * kH + j;
        if (gate == 0) {
          zb[o] = sigmoidf_(c + bias0[j]);                 // z
        } else {
          const float r = sigmoidf_(c + bias1[j]);         // r
          o16[o] = (f16)(r * hf[o]);                       // r*h for phase 2
        }
      } else {
        const size_t o = (size_t)m * kH + n;
        const float hc = tanhf(c + bias0[n]);
        const float z  = zb[o];
        const float h  = hf[o];
        const float hn = (1.0f - z) * h + z * hc;
        hf[o]  = hn;
        o16[o] = (f16)hn;                                  // f16 mirror for next step
      }
    }
  }
}

// ---------------- launch ----------------
extern "C" void kernel_launch(void* const* d_in, const int* in_sizes, int n_in,
                              void* d_out, int out_size, void* d_ws, size_t ws_size,
                              hipStream_t stream) {
  const float* x  = (const float*)d_in[0];
  const float* W  = (const float*)d_in[1];
  const float* U  = (const float*)d_in[2];
  const float* Wz = (const float*)d_in[3];
  const float* Uz = (const float*)d_in[4];
  const float* Wr = (const float*)d_in[5];
  const float* Ur = (const float*)d_in[6];
  const float* bc = (const float*)d_in[7];
  const float* bz = (const float*)d_in[8];
  const float* br = (const float*)d_in[9];

  char* ws = (char*)d_ws;
  f16*   WzrT = (f16*)ws;   ws += (size_t)2 * kH * kKZR * sizeof(f16);  // 6.0 MB
  f16*   WcT  = (f16*)ws;   ws += (size_t)kH * kKZR * sizeof(f16);      // 3.0 MB
  float* hf   = (float*)ws; ws += (size_t)kB * kH * sizeof(float);      // 1.0 MB
  f16*   h16  = (f16*)ws;   ws += (size_t)kB * kH * sizeof(f16);        // 0.5 MB
  f16*   rh16 = (f16*)ws;   ws += (size_t)kB * kH * sizeof(f16);        // 0.5 MB
  float* zb   = (float*)ws;                                             // 1.0 MB

  pack_zr_kernel<<<(2 * kH * kKZR + 255) / 256, 256, 0, stream>>>(Wz, Uz, Wr, Ur, WzrT);
  pack_c_kernel<<<(kH * kKZR + 255) / 256, 256, 0, stream>>>(W, U, WcT);
  init_h_kernel<<<(kB * kH + 255) / 256, 256, 0, stream>>>(hf, h16);

  const dim3 grid1(2 * kH / BN, kB / BM);  // 32 x 4 = 128 WGs
  const dim3 grid2(kH / BN, kB / BM);      // 16 x 4 =  64 WGs
  for (int t = 0; t < kT; ++t) {
    gru_step_kernel<0><<<grid1, NTHREADS, 0, stream>>>(x, t, WzrT, h16, hf, zb, rh16, bz, br);
    gru_step_kernel<1><<<grid2, NTHREADS, 0, stream>>>(x, t, WcT, rh16, hf, zb, h16, bc, nullptr);
  }

  copy_out_kernel<<<(kB * kH + 255) / 256, 256, 0, stream>>>(hf, (float*)d_out);
}
